// EViT_90950227460120
// MI455X (gfx1250) — compile-verified, matrix operand order checked by reading, not run on hardware
//
#include <hip/hip_runtime.h>
#include <hip/hip_bf16.h>
#include <math.h>

// ---------------------------------------------------------------------------
// EViT (ViT-B + token pruning @ layer 3) forward for gfx1250 (MI455X).
// One batched LDS-staged WMMA GEMM (v_wmma_f32_16x16x32_f16) serves all
// dense + attention matmuls. Double-buffered LDS (ping-pong) overlaps
// next-tile global staging with current-tile WMMA; one barrier per k-step.
// ---------------------------------------------------------------------------

typedef __attribute__((ext_vector_type(16))) _Float16 v16h;
typedef __attribute__((ext_vector_type(4)))  _Float16 v4h;
typedef __attribute__((ext_vector_type(8)))  float    v8f;

namespace {
constexpr int kB    = 64;
constexpr int kD    = 768;
constexpr int kH    = 12;
constexpr int kHD   = 64;     // head dim
constexpr int kMLP  = 3072;
constexpr int kNCLS = 100;
constexpr int kNP   = 196;    // patches
constexpr int kN0   = 197;    // tokens pre-prune
constexpr int kN0P  = 208;    // padded attn row stride (16-aligned)
constexpr int kKTOP = 137;    // kept tokens
constexpr int kN1   = 139;    // tokens post-prune (cls + 137 + rst)
constexpr int kSEL  = 3;
constexpr int kNL   = 12;
}

// K index inside a 16-wide f16 fragment (A: lane=M, B: lane=N), per ISA layout.
__device__ __forceinline__ int wmma_k(int e, int half) {
  int v = e >> 1, p = e & 1;
  return (v < 4) ? (half * 8 + v * 2 + p) : (16 + half * 8 + (v - 4) * 2 + p);
}

__device__ __forceinline__ float gelu_exact(float x) {
  return 0.5f * x * (1.0f + erff(x * 0.70710678118654752f));
}

// ---------------------------------------------------------------------------
// Batched WMMA GEMM:
//   z = blockIdx.z ; b = z / innerH ; h = z % innerH
//   C[b,h][M,N] = epi( oscale * A[b,h][M,K] * B[b,h][K,N] + bias ) (+ Cin)
// ACT: 0 none, 1 exact GELU.  BT: B operand stored as [N,K].
// Block = 256 threads (8 waves), 64x64 tile; wave w owns subtiles
// (w>>2, w&3) and ((w>>2)+2, w&3), sharing one B fragment per k-step.
// LDS: ping-pong As[2][m][k], BsT[2][n][k] (fragment loads contiguous in k).
// ---------------------------------------------------------------------------
template <int ACT, bool BT>
__global__ void gemm_wmma_kernel(const float* __restrict__ A, int lda,
                                 size_t sAb, size_t sAh,
                                 const float* __restrict__ Bm, int ldb,
                                 size_t sBb, size_t sBh,
                                 const float* __restrict__ bias,
                                 const float* __restrict__ Cin,
                                 float* __restrict__ C, int ldc,
                                 size_t sCb, size_t sCh,
                                 int innerH, int M, int N, int K, float oscale) {
  __shared__ _Float16 As[2][64][32];
  __shared__ _Float16 BsT[2][64][32];

  const int z  = blockIdx.z;
  const int bb = z / innerH;
  const int hh = z % innerH;
  A  += (size_t)bb * sAb + (size_t)hh * sAh;
  Bm += (size_t)bb * sBb + (size_t)hh * sBh;
  const size_t coff = (size_t)bb * sCb + (size_t)hh * sCh;
  C += coff;
  if (Cin) Cin += coff;

  const int tid  = threadIdx.x;
  const int wave = tid >> 5;
  const int lane = tid & 31;
  const int half = lane >> 4;
  const int lm   = lane & 15;

  const int m0 = blockIdx.y * 64;
  const int n0 = blockIdx.x * 64;

  const int smA = wave >> 2;        // first subtile row (0..1); second is +2
  const int sn  = wave & 3;         // subtile col (0..3)

  const bool vecA = ((lda & 3) == 0) &&
                    ((reinterpret_cast<uintptr_t>(A) & 15u) == 0);
  const bool vecB = ((ldb & 3) == 0) &&
                    ((reinterpret_cast<uintptr_t>(Bm) & 15u) == 0);

  // ---- staging lambdas (write one 64x32 k-tile into the given buffer) ----
  auto stageA = [&](int k0, _Float16 (*AsB)[32]) {
    if (m0 + 64 <= M && k0 + 32 <= K) {
      if (vecA) {
        #pragma unroll
        for (int i = tid; i < 512; i += 256) {          // 64 rows x 8 float4
          int r = i >> 3, c4 = (i & 7) << 2;
          const float4 f = *reinterpret_cast<const float4*>(
              A + (size_t)(m0 + r) * lda + (k0 + c4));
          v4h hv = {(_Float16)f.x, (_Float16)f.y, (_Float16)f.z, (_Float16)f.w};
          *reinterpret_cast<v4h*>(&AsB[r][c4]) = hv;
        }
        if (k0 + 32 < K)  // hint the tile after next into cache
          __builtin_prefetch(A + (size_t)(m0 + (tid >> 2)) * lda + (k0 + 32), 0, 1);
      } else {
        #pragma unroll
        for (int i = tid; i < 2048; i += 256) {
          int r = i >> 5, c = i & 31;
          AsB[r][c] = (_Float16)A[(size_t)(m0 + r) * lda + (k0 + c)];
        }
      }
    } else {
      #pragma unroll
      for (int i = tid; i < 2048; i += 256) {
        int r = i >> 5, c = i & 31;
        int gm = m0 + r, gk = k0 + c;
        float v = (gm < M && gk < K) ? A[(size_t)gm * lda + gk] : 0.0f;
        AsB[r][c] = (_Float16)v;
      }
    }
  };

  auto stageB = [&](int k0, _Float16 (*BsB)[32]) {
    if (n0 + 64 <= N && k0 + 32 <= K) {
      if (BT) {
        if (vecB) {
          #pragma unroll
          for (int i = tid; i < 512; i += 256) {        // 64 cols x 8 float4(k)
            int c = i >> 3, r4 = (i & 7) << 2;
            const float4 f = *reinterpret_cast<const float4*>(
                Bm + (size_t)(n0 + c) * ldb + (k0 + r4));
            v4h hv = {(_Float16)f.x, (_Float16)f.y, (_Float16)f.z, (_Float16)f.w};
            *reinterpret_cast<v4h*>(&BsB[c][r4]) = hv;
          }
        } else {
          #pragma unroll
          for (int i = tid; i < 2048; i += 256) {
            int c = i >> 5, r = i & 31;
            BsB[c][r] = (_Float16)Bm[(size_t)(n0 + c) * ldb + (k0 + r)];
          }
        }
      } else {
        if (vecB) {
          #pragma unroll
          for (int i = tid; i < 512; i += 256) {        // 32 rows x 16 float4(n)
            int r = i >> 4, c4 = (i & 15) << 2;
            const float4 f = *reinterpret_cast<const float4*>(
                Bm + (size_t)(k0 + r) * ldb + (n0 + c4));
            BsB[c4 + 0][r] = (_Float16)f.x;
            BsB[c4 + 1][r] = (_Float16)f.y;
            BsB[c4 + 2][r] = (_Float16)f.z;
            BsB[c4 + 3][r] = (_Float16)f.w;
          }
        } else {
          #pragma unroll
          for (int i = tid; i < 2048; i += 256) {
            int r = i >> 6, c = i & 63;
            BsB[c][r] = (_Float16)Bm[(size_t)(k0 + r) * ldb + (n0 + c)];
          }
        }
      }
    } else {
      #pragma unroll
      for (int i = tid; i < 2048; i += 256) {
        int c = i >> 5, r = i & 31;
        int gk = k0 + r, gn = n0 + c;
        float v = 0.0f;
        if (gk < K && gn < N)
          v = BT ? Bm[(size_t)gn * ldb + gk] : Bm[(size_t)gk * ldb + gn];
        BsB[c][r] = (_Float16)v;
      }
    }
  };

  v8f acc0 = {};
  v8f acc1 = {};

  // ---- pipelined main loop: stage k+32 into buf^1 while computing buf ----
  stageA(0, As[0]);
  stageB(0, BsT[0]);
  __syncthreads();

  int buf = 0;
  for (int k0 = 0; k0 < K; k0 += 32) {
    const int kn = k0 + 32;
    if (kn < K) {
      stageA(kn, As[buf ^ 1]);
      stageB(kn, BsT[buf ^ 1]);
    }

    v16h bfrag, a0, a1;
    #pragma unroll
    for (int e = 0; e < 16; ++e) {
      int kk = wmma_k(e, half);
      bfrag[e] = BsT[buf][sn * 16 + lm][kk];
      a0[e]    = As[buf][smA * 16 + lm][kk];
      a1[e]    = As[buf][(smA + 2) * 16 + lm][kk];
    }
    acc0 = __builtin_amdgcn_wmma_f32_16x16x32_f16(false, a0, false, bfrag,
                                                  (short)0, acc0, false, false);
    acc1 = __builtin_amdgcn_wmma_f32_16x16x32_f16(false, a1, false, bfrag,
                                                  (short)0, acc1, false, false);
    __syncthreads();
    buf ^= 1;
  }

  const int colg = n0 + sn * 16 + lm;
  if (colg < N) {
    const float bv = bias ? bias[colg] : 0.0f;
    #pragma unroll
    for (int r = 0; r < 8; ++r) {
      int rowg = m0 + smA * 16 + r + half * 8;
      if (rowg < M) {
        float v = acc0[r] * oscale + bv;
        if (ACT == 1) v = gelu_exact(v);
        if (Cin) v += Cin[(size_t)rowg * ldc + colg];
        C[(size_t)rowg * ldc + colg] = v;
      }
      rowg = m0 + (smA + 2) * 16 + r + half * 8;
      if (rowg < M) {
        float v = acc1[r] * oscale + bv;
        if (ACT == 1) v = gelu_exact(v);
        if (Cin) v += Cin[(size_t)rowg * ldc + colg];
        C[(size_t)rowg * ldc + colg] = v;
      }
    }
  }
}

// ---------------------------------------------------------------------------
// im2col: [B,3,224,224] -> [B*196, 768] with col = c*256 + py*16 + px
// ---------------------------------------------------------------------------
__global__ void im2col_kernel(const float* __restrict__ in, float* __restrict__ out) {
  size_t i = (size_t)blockIdx.x * 256 + threadIdx.x;
  if (i >= (size_t)kB * kNP * kD) return;
  int d = (int)(i % kD);
  size_t t = i / kD;
  int p = (int)(t % kNP);
  int b = (int)(t / kNP);
  int c  = d >> 8;
  int py = (d >> 4) & 15;
  int px = d & 15;
  int ph = p / 14, pw = p % 14;
  int y = ph * 16 + py, x = pw * 16 + px;
  out[i] = in[(((size_t)b * 3 + c) * 224 + y) * 224 + x];
}

// x[b,0,:] = cls + pos[0];  x[b,1+i,:] = patch[b*196+i,:] + pos[1+i]
__global__ void assemble_x_kernel(const float* __restrict__ patch,
                                  const float* __restrict__ cls,
                                  const float* __restrict__ pos,
                                  float* __restrict__ x) {
  size_t i = (size_t)blockIdx.x * 256 + threadIdx.x;
  if (i >= (size_t)kB * kN0 * kD) return;
  int d = (int)(i % kD);
  size_t t = i / kD;
  int n = (int)(t % kN0);
  int b = (int)(t / kN0);
  float v;
  if (n == 0) v = cls[d];
  else        v = patch[((size_t)b * kNP + (n - 1)) * kD + d];
  x[i] = v + pos[(size_t)n * kD + d];
}

// LayerNorm over 768 cols; one 256-thread block per row.
__global__ void layernorm_kernel(const float* __restrict__ x,
                                 const float* __restrict__ g,
                                 const float* __restrict__ bta,
                                 float* __restrict__ y) {
  __shared__ float red[256];
  const float* row = x + (size_t)blockIdx.x * kD;
  float* orow = y + (size_t)blockIdx.x * kD;
  int t = threadIdx.x;
  float v0 = row[t], v1 = row[t + 256], v2 = row[t + 512];
  red[t] = v0 + v1 + v2;
  __syncthreads();
  for (int s = 128; s > 0; s >>= 1) { if (t < s) red[t] += red[t + s]; __syncthreads(); }
  float mean = red[0] * (1.0f / kD);
  __syncthreads();
  float d0 = v0 - mean, d1 = v1 - mean, d2 = v2 - mean;
  red[t] = d0 * d0 + d1 * d1 + d2 * d2;
  __syncthreads();
  for (int s = 128; s > 0; s >>= 1) { if (t < s) red[t] += red[t + s]; __syncthreads(); }
  float rstd = rsqrtf(red[0] * (1.0f / kD) + 1e-6f);
  orow[t]       = d0 * rstd * g[t]       + bta[t];
  orow[t + 256] = d1 * rstd * g[t + 256] + bta[t + 256];
  orow[t + 512] = d2 * rstd * g[t + 512] + bta[t + 512];
}

// Softmax over row of length Nt, stored with stride NtP; one block per row.
__global__ void softmax_kernel(float* __restrict__ attn, int Nt, int NtP) {
  __shared__ float red[256];
  float* p = attn + (size_t)blockIdx.x * NtP;
  int t = threadIdx.x;
  float v = (t < Nt) ? p[t] : -3.4e38f;
  red[t] = v;
  __syncthreads();
  for (int s = 128; s > 0; s >>= 1) { if (t < s) red[t] = fmaxf(red[t], red[t + s]); __syncthreads(); }
  float m = red[0];
  __syncthreads();
  float e = (t < Nt) ? __expf(v - m) : 0.0f;
  red[t] = e;
  __syncthreads();
  for (int s = 128; s > 0; s >>= 1) { if (t < s) red[t] += red[t + s]; __syncthreads(); }
  float inv = 1.0f / red[0];
  if (t < Nt) p[t] = e * inv;
}

// importance[b, j] = mean_h attn[b,h,0,1+j]   (attn rows strided by NtP)
__global__ void importance_kernel(const float* __restrict__ attn,
                                  float* __restrict__ imp, int Nt, int NtP) {
  int b = blockIdx.x, j = threadIdx.x;
  if (j >= kNP) return;
  float s = 0.0f;
  #pragma unroll
  for (int h = 0; h < kH; ++h)
    s += attn[(size_t)(b * kH + h) * Nt * NtP + 1 + j];
  imp[(size_t)b * kNP + j] = s * (1.0f / kH);
}

// Top-137 indices in descending value order (first-index tie break, as top_k).
__global__ void topk_kernel(const float* __restrict__ imp, int* __restrict__ idx) {
  int b = blockIdx.x;
  float vals[kNP];
  const float* p = imp + (size_t)b * kNP;
  for (int i = 0; i < kNP; ++i) vals[i] = p[i];
  for (int t = 0; t < kKTOP; ++t) {
    int best = 0; float bv = vals[0];
    for (int i = 1; i < kNP; ++i) if (vals[i] > bv) { bv = vals[i]; best = i; }
    idx[b * kKTOP + t] = best;
    vals[best] = -3.4e38f;
  }
}

// x2[b] = concat(cls, gathered 137 tokens, rst);  rst = sum(all) - sum(kept)
__global__ void gather_prune_kernel(const float* __restrict__ x,
                                    const int* __restrict__ idx,
                                    float* __restrict__ x2) {
  int b = blockIdx.x;
  int d = blockIdx.y * 256 + threadIdx.x;
  if (d >= kD) return;
  const float* xb = x + (size_t)b * kN0 * kD;
  float* yb = x2 + (size_t)b * kN1 * kD;
  yb[d] = xb[d];  // cls
  float total = 0.0f;
  for (int i = 0; i < kNP; ++i) total += xb[(size_t)(1 + i) * kD + d];
  float sel = 0.0f;
  for (int t = 0; t < kKTOP; ++t) {
    int id = idx[b * kKTOP + t];
    float v = xb[(size_t)(1 + id) * kD + d];
    yb[(size_t)(1 + t) * kD + d] = v;
    sel += v;
  }
  yb[(size_t)(kN1 - 1) * kD + d] = total - sel;
}

// ---------------------------------------------------------------------------
// Host orchestration
// ---------------------------------------------------------------------------
static void launch_gemm_ex(int act, bool bt,
                           const float* A, int lda, size_t sAb, size_t sAh,
                           const float* Bm, int ldb, size_t sBb, size_t sBh,
                           const float* bias, const float* Cin,
                           float* C, int ldc, size_t sCb, size_t sCh,
                           int innerH, int nbatch, int M, int N, int K,
                           float oscale, hipStream_t s) {
  dim3 grid((N + 63) / 64, (M + 63) / 64, nbatch), blk(256);
  if (bt)
    gemm_wmma_kernel<0, true><<<grid, blk, 0, s>>>(A, lda, sAb, sAh, Bm, ldb, sBb, sBh,
                                                   bias, Cin, C, ldc, sCb, sCh,
                                                   innerH, M, N, K, oscale);
  else if (act == 1)
    gemm_wmma_kernel<1, false><<<grid, blk, 0, s>>>(A, lda, sAb, sAh, Bm, ldb, sBb, sBh,
                                                    bias, Cin, C, ldc, sCb, sCh,
                                                    innerH, M, N, K, oscale);
  else
    gemm_wmma_kernel<0, false><<<grid, blk, 0, s>>>(A, lda, sAb, sAh, Bm, ldb, sBb, sBh,
                                                    bias, Cin, C, ldc, sCb, sCh,
                                                    innerH, M, N, K, oscale);
}

static void launch_gemm(int act, bool bt,
                        const float* A, int lda, const float* Bm, int ldb,
                        const float* bias, const float* Cin,
                        float* C, int ldc, int M, int N, int K, hipStream_t s) {
  launch_gemm_ex(act, bt, A, lda, 0, 0, Bm, ldb, 0, 0, bias, Cin,
                 C, ldc, 0, 0, 1, 1, M, N, K, 1.0f, s);
}

extern "C" void kernel_launch(void* const* d_in, const int* in_sizes, int n_in,
                              void* d_out, int out_size, void* d_ws, size_t ws_size,
                              hipStream_t stream) {
  (void)in_sizes; (void)n_in; (void)out_size; (void)ws_size;
  auto P = [&](int i) { return (const float*)d_in[i]; };

  const float* inputs  = P(0);
  const float* patch_w = P(1);
  const float* patch_b = P(2);
  const float* cls     = P(3);
  const float* pos     = P(4);
  const float* norm_g  = P(149);
  const float* norm_b  = P(150);
  const float* head_w  = P(151);
  const float* head_b  = P(152);

  // ---- workspace layout (floats) ----
  float* ws = (float*)d_ws;
  size_t o = 0;
  float* x    = ws + o; o += (size_t)kB * kN0 * kD;
  float* x2   = ws + o; o += (size_t)kB * kN1 * kD;
  float* h    = ws + o; o += (size_t)kB * kN0 * kD;
  float* qkv  = ws + o; o += (size_t)kB * kN0 * 3 * kD;
  float* attn = ws + o; o += (size_t)kB * kH * kN0 * kN0P;  // padded rows
  float* aout = ws + o; o += (size_t)kB * kN0 * kD;
  float* mlp  = ws + o; o += (size_t)kB * kN0 * kMLP;   // also reused for patch staging
  float* imp  = ws + o; o += (size_t)kB * kNP;
  int*   idx  = (int*)(ws + o);

  // ---- patch embed: im2col + GEMM (Bt, since patch_w is [D, 768]) ----
  float* patches  = mlp;                                  // [B*196, 768]
  float* patchout = mlp + (size_t)kB * kNP * kD;          // [B*196, 768]
  {
    size_t tot = (size_t)kB * kNP * kD;
    im2col_kernel<<<(unsigned)((tot + 255) / 256), 256, 0, stream>>>(inputs, patches);
    launch_gemm(0, true, patches, kD, patch_w, kD, patch_b, nullptr,
                patchout, kD, kB * kNP, kD, kD, stream);
    size_t totx = (size_t)kB * kN0 * kD;
    assemble_x_kernel<<<(unsigned)((totx + 255) / 256), 256, 0, stream>>>(patchout, cls, pos, x);
  }

  const float scale = 0.125f;  // hd^-0.5 = 1/8

  for (int l = 0; l < kNL; ++l) {
    const int base = 5 + l * 12;
    const float* ln1_g = P(base + 0);
    const float* ln1_b = P(base + 1);
    const float* qkv_w = P(base + 2);
    const float* qkv_b = P(base + 3);
    const float* prj_w = P(base + 4);
    const float* prj_b = P(base + 5);
    const float* ln2_g = P(base + 6);
    const float* ln2_b = P(base + 7);
    const float* fc1_w = P(base + 8);
    const float* fc1_b = P(base + 9);
    const float* fc2_w = P(base + 10);
    const float* fc2_b = P(base + 11);

    int Nt  = (l <= kSEL) ? kN0 : kN1;
    int NtP = (Nt + 15) & ~15;                 // padded attn row stride
    float* xc = (l <= kSEL) ? x : x2;
    int M = kB * Nt;

    // --- attention ---
    layernorm_kernel<<<M, 256, 0, stream>>>(xc, ln1_g, ln1_b, h);
    launch_gemm(0, false, h, kD, qkv_w, 3 * kD, qkv_b, nullptr,
                qkv, 3 * kD, M, 3 * kD, kD, stream);

    // scores: S[b,h] = (Q K^T) * scale    [batched over B*H, C rows padded]
    launch_gemm_ex(0, true,
                   /*A=Q*/ qkv, 3 * kD, (size_t)Nt * 3 * kD, (size_t)kHD,
                   /*B=K*/ qkv + kD, 3 * kD, (size_t)Nt * 3 * kD, (size_t)kHD,
                   nullptr, nullptr,
                   /*C*/ attn, NtP, (size_t)kH * Nt * NtP, (size_t)Nt * NtP,
                   kH, kB * kH, Nt, Nt, kHD, scale, stream);

    softmax_kernel<<<kB * kH * Nt, 256, 0, stream>>>(attn, Nt, NtP);
    if (l == kSEL)
      importance_kernel<<<kB, 256, 0, stream>>>(attn, imp, Nt, NtP);

    // O[b,h] = attn @ V  -> aout[b, n, h*64+d]            [batched over B*H]
    launch_gemm_ex(0, false,
                   /*A=attn*/ attn, NtP, (size_t)kH * Nt * NtP, (size_t)Nt * NtP,
                   /*B=V*/ qkv + 2 * kD, 3 * kD, (size_t)Nt * 3 * kD, (size_t)kHD,
                   nullptr, nullptr,
                   /*C*/ aout, kD, (size_t)Nt * kD, (size_t)kHD,
                   kH, kB * kH, Nt, kHD, Nt, 1.0f, stream);

    // proj + residual (in place: each output element read+written by one thread)
    launch_gemm(0, false, aout, kD, prj_w, kD, prj_b, xc, xc, kD, M, kD, kD, stream);

    // --- token pruning at layer 3 (after attention residual) ---
    if (l == kSEL) {
      topk_kernel<<<kB, 1, 0, stream>>>(imp, idx);
      gather_prune_kernel<<<dim3(kB, (kD + 255) / 256), 256, 0, stream>>>(x, idx, x2);
      xc = x2;
      Nt = kN1;
      M = kB * kN1;
    }

    // --- MLP ---
    layernorm_kernel<<<M, 256, 0, stream>>>(xc, ln2_g, ln2_b, h);
    launch_gemm(1, false, h, kD, fc1_w, kMLP, fc1_b, nullptr,
                mlp, kMLP, M, kMLP, kD, stream);
    launch_gemm(0, false, mlp, kMLP, fc2_w, kD, fc2_b, xc, xc, kD, M, kD, kMLP, stream);
  }

  // ---- final LN + head -> d_out [B, 139, 100] ----
  layernorm_kernel<<<kB * kN1, 256, 0, stream>>>(x2, norm_g, norm_b, h);
  launch_gemm(0, false, h, kD, head_w, kNCLS, head_b, nullptr,
              (float*)d_out, kNCLS, kB * kN1, kNCLS, kD, stream);
}
